// LFA_54846732370506
// MI455X (gfx1250) — matrix operand order repaired and checked by reading, not compiled, not run on hardware
//
#include <hip/hip_runtime.h>

// Problem constants (from reference setup_inputs)
#define BB 2
#define NN 8192
#define CIN 64
#define CCAT 128
#define COUT 128

typedef __attribute__((ext_vector_type(16))) __bf16 v16bf;
typedef __attribute__((ext_vector_type(8)))  __bf16 v8bf;
typedef __attribute__((ext_vector_type(8)))  float  v8f;

static __device__ __forceinline__ float lrelu(float x) { return x >= 0.0f ? x : 0.2f * x; }

static __device__ __forceinline__ float pdist3(const float* __restrict__ xb, int i, int j) {
    float dx = xb[i * 3 + 0] - xb[j * 3 + 0];
    float dy = xb[i * 3 + 1] - xb[j * 3 + 1];
    float dz = xb[i * 3 + 2] - xb[j * 3 + 2];
    return sqrtf(fmaxf(dx * dx + dy * dy + dz * dz, 1e-12f));
}

// ---------------------------------------------------------------------------
// Weight f32 -> bf16 staging
// ---------------------------------------------------------------------------
__global__ void cvt_bf16_kernel(const float* __restrict__ s, __bf16* __restrict__ d, int n) {
    int i = blockIdx.x * 256 + threadIdx.x;
    if (i < n) d[i] = (__bf16)s[i];
}

// ---------------------------------------------------------------------------
// Brute-force KNN (top-4 incl. self) with LDS column tiling.
// One thread per query point; strict '<' keeps earliest index on ties.
// ---------------------------------------------------------------------------
__global__ __launch_bounds__(256)
void knn_kernel(const float* __restrict__ xyz, int* __restrict__ idxb, float* __restrict__ nd) {
    __shared__ float sx[256], sy[256], sz[256];
    const int b = blockIdx.y;
    const int q = blockIdx.x * 256 + threadIdx.x;
    const float* xb = xyz + (size_t)b * NN * 3;
    const float qx = xb[q * 3 + 0], qy = xb[q * 3 + 1], qz = xb[q * 3 + 2];

    float d0 = 3.4e38f, d1 = 3.4e38f, d2v = 3.4e38f, d3 = 3.4e38f;
    int   i0 = 0, i1 = 0, i2 = 0, i3 = 0;

    for (int j0 = 0; j0 < NN; j0 += 256) {
        int j = j0 + threadIdx.x;
        sx[threadIdx.x] = xb[j * 3 + 0];
        sy[threadIdx.x] = xb[j * 3 + 1];
        sz[threadIdx.x] = xb[j * 3 + 2];
        __syncthreads();
        for (int t = 0; t < 256; ++t) {
            float dx = qx - sx[t], dy = qy - sy[t], dz = qz - sz[t];
            float d = sqrtf(fmaxf(dx * dx + dy * dy + dz * dz, 1e-12f));
            int jj = j0 + t;
            if (d < d3) {
                if (d < d2v) {
                    d3 = d2v; i3 = i2;
                    if (d < d1) {
                        d2v = d1; i2 = i1;
                        if (d < d0) { d1 = d0; i1 = i0; d0 = d; i0 = jj; }
                        else        { d1 = d;  i1 = jj; }
                    } else { d2v = d; i2 = jj; }
                } else { d3 = d; i3 = jj; }
            }
        }
        __syncthreads();
    }
    size_t base = ((size_t)b * NN + q) * 4;
    idxb[base + 0] = i0; idxb[base + 1] = i1; idxb[base + 2] = i2; idxb[base + 3] = i3;
    nd[base + 0] = d0;   nd[base + 1] = d1;   nd[base + 2] = d2v;  nd[base + 3] = d3;
}

// ---------------------------------------------------------------------------
// intra = [nd1, nd2, nd3, d(i1,i2), d(i2,i3), d(i3,i1)]
// ---------------------------------------------------------------------------
__global__ __launch_bounds__(256)
void intra_kernel(const float* __restrict__ xyz, const int* __restrict__ idxb,
                  const float* __restrict__ nd, float* __restrict__ intra) {
    const int b = blockIdx.y;
    const int n = blockIdx.x * 256 + threadIdx.x;
    const float* xb = xyz + (size_t)b * NN * 3;
    size_t base = ((size_t)b * NN + n) * 4;
    int i1 = idxb[base + 1], i2 = idxb[base + 2], i3 = idxb[base + 3];
    float* o = intra + ((size_t)b * NN + n) * 6;
    o[0] = nd[base + 1];
    o[1] = nd[base + 2];
    o[2] = nd[base + 3];
    o[3] = pdist3(xb, i1, i2);
    o[4] = pdist3(xb, i2, i3);
    o[5] = pdist3(xb, i3, i1);
}

// ---------------------------------------------------------------------------
// rel_raw[b,n,k,:] = [intra[n] (6) | intra[idx[n,k]] (6) | inter (4)]
// inter[a] = d(xyz[idx[n,a]], xyz[idx[idx[n,k],a]])
// ---------------------------------------------------------------------------
__global__ __launch_bounds__(256)
void relraw_kernel(const float* __restrict__ xyz, const int* __restrict__ idxb,
                   const float* __restrict__ intra, float* __restrict__ rel) {
    const int b = blockIdx.y;
    const int g = blockIdx.x * 256 + threadIdx.x;   // n*4 + k
    const int n = g >> 2, k = g & 3;
    const float* xb = xyz + (size_t)b * NN * 3;
    size_t nb = (size_t)b * NN + n;
    int j = idxb[nb * 4 + k];
    size_t jb = (size_t)b * NN + j;

    float row[16];
    const float* ci = intra + nb * 6;
    const float* ni = intra + jb * 6;
#pragma unroll
    for (int a = 0; a < 6; ++a) { row[a] = ci[a]; row[6 + a] = ni[a]; }
#pragma unroll
    for (int a = 0; a < 4; ++a) {
        int ia = idxb[nb * 4 + a];
        int ja = idxb[jb * 4 + a];
        row[12 + a] = pdist3(xb, ia, ja);
    }
    float* o = rel + (nb * 4 + k) * 16;
#pragma unroll
    for (int a = 0; a < 16; ++a) o[a] = row[a];
}

// ---------------------------------------------------------------------------
// WMMA fragment loaders (ISA 7.12.2 16-bit layouts, wave32)
// A 16x32: lane holds row (lane&15); K = k0 + 8*half + {0..7}, +16 + {0..7}
// B 32x16: lane holds col (lane&15); K = k0 + 8*half + {0..7}, +16 + {0..7}
// ---------------------------------------------------------------------------
static __device__ __forceinline__ v16bf fragA_lds(const __bf16* base, int row, int ld, int k0, int h) {
    const __bf16* p = base + row * ld + k0 + 8 * h;
    v8bf lo = *(const v8bf*)(p);
    v8bf hi = *(const v8bf*)(p + 16);
    v16bf r;
#pragma unroll
    for (int i = 0; i < 8; ++i) { r[i] = lo[i]; r[i + 8] = hi[i]; }
    return r;
}

static __device__ __forceinline__ v16bf fragB_g(const __bf16* __restrict__ W, int c, int ld, int k0, int h) {
    const __bf16* p = W + (size_t)c * ld + k0 + 8 * h;
    v8bf lo = *(const v8bf*)(p);
    v8bf hi = *(const v8bf*)(p + 16);
    v16bf r;
#pragma unroll
    for (int i = 0; i < 8; ++i) { r[i] = lo[i]; r[i + 8] = hi[i]; }
    return r;
}

// ---------------------------------------------------------------------------
// Fused: rel MLP -> feat assembly -> attn GEMM (WMMA) -> in-lane softmax/pool
//        -> W_out & W_sc GEMMs (WMMA) -> BN + LeakyReLU epilogue.
// Block = 32 points (128 (n,k) rows), 256 threads = 8 waves.
// ---------------------------------------------------------------------------
__global__ __launch_bounds__(256)
void lfa_fused(const float* __restrict__ feature,
               const int*   __restrict__ idxb,
               const float* __restrict__ relraw,
               const float* __restrict__ W_rel,  const float* __restrict__ b_rel,
               const float* __restrict__ g_rel,  const float* __restrict__ be_rel,
               const __bf16* __restrict__ Wattn,
               const __bf16* __restrict__ Wout,
               const __bf16* __restrict__ Wsc,
               const float* __restrict__ b_out, const float* __restrict__ g_out, const float* __restrict__ be_out,
               const float* __restrict__ b_sc,  const float* __restrict__ g_sc,  const float* __restrict__ be_sc,
               float* __restrict__ outp) {
    __shared__ __align__(16) __bf16 sFeat[128 * 128];   // 32 KB: (n,k) rows x Ccat
    __shared__ __align__(16) __bf16 sPooled[32 * 128];  //  8 KB
    __shared__ __align__(16) __bf16 sFeatA[32 * 64];    //  4 KB: raw feature tile

    const int tid = threadIdx.x;
    const int b   = blockIdx.y;
    const int n0  = blockIdx.x * 32;
    const float* fb = feature + (size_t)b * NN * CIN;
    const float bninv = rsqrtf(1.0f + 1e-5f);

    __builtin_prefetch(Wattn, 0, 3);
    __builtin_prefetch(Wout, 0, 3);
    __builtin_prefetch(Wsc, 0, 3);

    // ---- stage 1: build sFeat = [gathered features | rel MLP], and sFeatA ----
    {
        const int r    = tid >> 1;      // row 0..127 = (point-in-block)*4 + k
        const int half = tid & 1;
        const int pt   = r >> 2;
        const int k    = r & 3;
        const int j    = idxb[(((size_t)b * NN + n0 + pt) << 2) + k];

        // cols [half*32, half*32+32): gathered neighbor features
        const float* fr = fb + (size_t)j * CIN + half * 32;
#pragma unroll
        for (int c = 0; c < 32; ++c)
            sFeat[r * 128 + half * 32 + c] = (__bf16)fr[c];

        // cols 64 + [half*32, half*32+32): rel = lrelu(bn(rel_raw @ W_rel^T + b_rel))
        const float4* rp = (const float4*)(relraw + ((size_t)(b * NN + n0)) * 64 + (size_t)r * 16);
        float4 t0 = rp[0], t1 = rp[1], t2 = rp[2], t3 = rp[3];
        float rr[16] = { t0.x, t0.y, t0.z, t0.w, t1.x, t1.y, t1.z, t1.w,
                         t2.x, t2.y, t2.z, t2.w, t3.x, t3.y, t3.z, t3.w };
        const int cb = half * 32;
#pragma unroll 4
        for (int c = 0; c < 32; ++c) {
            int co = cb + c;
            float a = b_rel[co];
            const float* wr = W_rel + co * 16;
#pragma unroll
            for (int jx = 0; jx < 16; ++jx) a += rr[jx] * wr[jx];
            float v = a * (g_rel[co] * bninv) + be_rel[co];
            sFeat[r * 128 + 64 + co] = (__bf16)lrelu(v);
        }

        // raw feature tile for the shortcut GEMM
#pragma unroll
        for (int i = 0; i < 8; ++i) {
            int lin = tid * 8 + i;          // 0..2047
            int p = lin >> 6, c = lin & 63;
            sFeatA[lin] = (__bf16)fb[(size_t)(n0 + p) * CIN + c];
        }
    }
    __syncthreads();

    // ---- stage 2: attn logits GEMM + in-lane softmax over K + attention pooling ----
    const int w  = tid >> 5;        // wave id 0..7 == M-tile
    const int l  = tid & 31;
    const int h  = l >> 4;
    const int lc = l & 15;
    const int arow = 16 * w + lc;   // A-fragment row

    for (int nt = 0; nt < 8; ++nt) {
        v8f acc = {0.f, 0.f, 0.f, 0.f, 0.f, 0.f, 0.f, 0.f};
#pragma unroll
        for (int kk = 0; kk < 4; ++kk) {
            v16bf a  = fragA_lds(sFeat, arow, 128, 32 * kk, h);
            v16bf bb = fragB_g(Wattn, 16 * nt + lc, 128, 32 * kk, h);
            acc = __builtin_amdgcn_wmma_f32_16x16x32_bf16(false, a, false, bb, (short)0, acc, false, false);
        }
        const int col = 16 * nt + lc;
        // group 0: acc[0..3] = rows (8h+0..3) -> point 4w+2h, k=0..3
        {
            float a0 = acc[0], a1 = acc[1], a2 = acc[2], a3 = acc[3];
            float mx = fmaxf(fmaxf(a0, a1), fmaxf(a2, a3));
            float e0 = __expf(a0 - mx), e1 = __expf(a1 - mx), e2 = __expf(a2 - mx), e3 = __expf(a3 - mx);
            float s = e0 + e1 + e2 + e3;
            int pA = 4 * w + 2 * h;
            int fbase = (4 * pA) * 128 + col;
            float f0 = (float)sFeat[fbase + 0 * 128];
            float f1 = (float)sFeat[fbase + 1 * 128];
            float f2 = (float)sFeat[fbase + 2 * 128];
            float f3 = (float)sFeat[fbase + 3 * 128];
            sPooled[pA * 128 + col] = (__bf16)((e0 * f0 + e1 * f1 + e2 * f2 + e3 * f3) / s);
        }
        // group 1: acc[4..7] -> point 4w+2h+1
        {
            float a0 = acc[4], a1 = acc[5], a2 = acc[6], a3 = acc[7];
            float mx = fmaxf(fmaxf(a0, a1), fmaxf(a2, a3));
            float e0 = __expf(a0 - mx), e1 = __expf(a1 - mx), e2 = __expf(a2 - mx), e3 = __expf(a3 - mx);
            float s = e0 + e1 + e2 + e3;
            int pB = 4 * w + 2 * h + 1;
            int fbase = (4 * pB) * 128 + col;
            float f0 = (float)sFeat[fbase + 0 * 128];
            float f1 = (float)sFeat[fbase + 1 * 128];
            float f2 = (float)sFeat[fbase + 2 * 128];
            float f3 = (float)sFeat[fbase + 3 * 128];
            sPooled[pB * 128 + col] = (__bf16)((e0 * f0 + e1 * f1 + e2 * f2 + e3 * f3) / s);
        }
    }
    __syncthreads();

    // ---- stage 3: out = lrelu(bn(feature@W_sc^T + b_sc) + bn(pooled@W_out^T + b_out)) ----
    const int c = 16 * w + lc;                 // wave w owns output-channel tile w
    const float gsO = g_out[c] * bninv, beo = be_out[c], bo = b_out[c];
    const float gsS = g_sc[c]  * bninv, bes = be_sc[c],  bs = b_sc[c];

#pragma unroll
    for (int mt = 0; mt < 2; ++mt) {
        v8f accO = {0.f, 0.f, 0.f, 0.f, 0.f, 0.f, 0.f, 0.f};
#pragma unroll
        for (int kk = 0; kk < 4; ++kk) {
            v16bf a  = fragA_lds(sPooled, 16 * mt + lc, 128, 32 * kk, h);
            v16bf bb = fragB_g(Wout, c, 128, 32 * kk, h);
            accO = __builtin_amdgcn_wmma_f32_16x16x32_bf16(false, a, false, bb, (short)0, accO, false, false);
        }
        v8f accS = {0.f, 0.f, 0.f, 0.f, 0.f, 0.f, 0.f, 0.f};
#pragma unroll
        for (int kk = 0; kk < 2; ++kk) {
            v16bf a  = fragA_lds(sFeatA, 16 * mt + lc, 64, 32 * kk, h);
            v16bf bb = fragB_g(Wsc, c, 64, 32 * kk, h);
            accS = __builtin_amdgcn_wmma_f32_16x16x32_bf16(false, a, false, bb, (short)0, accS, false, false);
        }
#pragma unroll
        for (int g = 0; g < 8; ++g) {
            int row = 16 * mt + 8 * h + g;     // point in block
            float v = ((accS[g] + bs) * gsS + bes) + ((accO[g] + bo) * gsO + beo);
            outp[((size_t)(b * NN + n0 + row)) * COUT + c] = lrelu(v);
        }
    }
}

// ---------------------------------------------------------------------------
// Host launcher
// ---------------------------------------------------------------------------
extern "C" void kernel_launch(void* const* d_in, const int* in_sizes, int n_in,
                              void* d_out, int out_size, void* d_ws, size_t ws_size,
                              hipStream_t stream) {
    (void)in_sizes; (void)n_in; (void)out_size; (void)ws_size;

    const float* xyz     = (const float*)d_in[0];
    const float* feature = (const float*)d_in[1];
    const float* W_rel   = (const float*)d_in[2];
    const float* b_rel   = (const float*)d_in[3];
    const float* g_rel   = (const float*)d_in[4];
    const float* be_rel  = (const float*)d_in[5];
    const float* W_attn  = (const float*)d_in[6];
    const float* W_out   = (const float*)d_in[7];
    const float* b_out   = (const float*)d_in[8];
    const float* g_out   = (const float*)d_in[9];
    const float* be_out  = (const float*)d_in[10];
    const float* W_sc    = (const float*)d_in[11];
    const float* b_sc    = (const float*)d_in[12];
    const float* g_sc    = (const float*)d_in[13];
    const float* be_sc   = (const float*)d_in[14];

    float* out = (float*)d_out;
    const size_t o_out = (size_t)BB * NN * 3;                   // after xyz
    const size_t o_rel = o_out + (size_t)BB * NN * COUT;        // after out
    const size_t o_idx = o_rel + (size_t)BB * NN * 4 * 16;      // after rel_raw

    char* ws = (char*)d_ws;
    float* nd      = (float*)ws;                                        // B*N*4 f32
    size_t c0      = (size_t)BB * NN * 4 * sizeof(float);
    float* intra   = (float*)(ws + c0);                                 // B*N*6 f32
    size_t c1      = c0 + (size_t)BB * NN * 6 * sizeof(float);
    __bf16* wattnB = (__bf16*)(ws + c1);                                // 128x128
    __bf16* woutB  = wattnB + 128 * 128;                                // 128x128
    __bf16* wscB   = woutB + 128 * 128;                                 // 128x64

    int*   idxb   = (int*)(out + o_idx);
    float* relbuf = out + o_rel;

    // output 0: xyz passthrough
    hipMemcpyAsync(out, xyz, (size_t)BB * NN * 3 * sizeof(float), hipMemcpyDeviceToDevice, stream);

    cvt_bf16_kernel<<<dim3((128 * 128 + 255) / 256), 256, 0, stream>>>(W_attn, wattnB, 128 * 128);
    cvt_bf16_kernel<<<dim3((128 * 128 + 255) / 256), 256, 0, stream>>>(W_out,  woutB,  128 * 128);
    cvt_bf16_kernel<<<dim3((128 * 64  + 255) / 256), 256, 0, stream>>>(W_sc,   wscB,   128 * 64);

    knn_kernel   <<<dim3(NN / 256, BB), 256, 0, stream>>>(xyz, idxb, nd);
    intra_kernel <<<dim3(NN / 256, BB), 256, 0, stream>>>(xyz, idxb, nd, intra);
    relraw_kernel<<<dim3(NN * 4 / 256, BB), 256, 0, stream>>>(xyz, idxb, intra, relbuf);

    lfa_fused<<<dim3(NN / 32, BB), 256, 0, stream>>>(
        feature, idxb, relbuf, W_rel, b_rel, g_rel, be_rel,
        wattnB, woutB, wscB,
        b_out, g_out, be_out, b_sc, g_sc, be_sc,
        out + o_out);
}
